// MultiHeadAttention_5377299054939
// MI455X (gfx1250) — matrix operand order repaired and checked by reading, not compiled, not run on hardware
//
#include <hip/hip_runtime.h>
#include <hip/hip_bf16.h>
#include <cstdint>
#include <cstddef>

// Problem dims (fixed by reference)
#define BB   4
#define SS   2048
#define DM   512
#define NH   8
#define DK   64
#define ROWS (BB * SS)          // 8192 token rows

typedef __attribute__((ext_vector_type(16))) _Float16 v16h;
typedef __attribute__((ext_vector_type(8)))  float    v8f;

union FragU { v16h v; float4 q[2]; };

__device__ __forceinline__ v8f wmma16x16x32(v16h a, v16h b, v8f c) {
  // D = A(16x32 f16) * B(32x16 f16) + C(16x16 f32)
  return __builtin_amdgcn_wmma_f32_16x16x32_f16(
      /*neg_a=*/false, a, /*neg_b=*/false, b,
      /*c_mod=*/(short)0, c, /*reuse_a=*/false, /*reuse_b=*/false);
}

__device__ __forceinline__ v8f zero_v8f() {
  v8f z;
#pragma unroll
  for (int i = 0; i < 8; ++i) z[i] = 0.0f;
  return z;
}

// A fragment (16x32 f16), ISA layout: lane m = L&15; lane-half hi selects k in
// [hi*8, hi*8+8) and [16+hi*8, ...). Two contiguous 16B loads per lane.
__device__ __forceinline__ v16h load_frag_a(const _Float16* __restrict__ base,
                                            int lda, int mBase, int kBase) {
  const int lane = threadIdx.x & 31;
  const int m = lane & 15, hi = lane >> 4;
  const _Float16* p = base + (size_t)(mBase + m) * lda + kBase;
  FragU f;
  f.q[0] = *reinterpret_cast<const float4*>(p + hi * 8);
  f.q[1] = *reinterpret_cast<const float4*>(p + 16 + hi * 8);
  return f.v;
}

// B fragment (32x16 f16) where element(k,n) = M[(nBase+n)*ldb + kBase + k]
// (matrix stored n-major / "transposed"). Lane n = L&15; lane-half hi holds k in
// [hi*8, hi*8+8) and [16+hi*8, ...) => two contiguous 16B loads.
__device__ __forceinline__ v16h load_frag_bT(const _Float16* __restrict__ base,
                                             int ldb, int nBase, int kBase) {
  const int lane = threadIdx.x & 31;
  const int n = lane & 15, hi = lane >> 4;
  const _Float16* p = base + (size_t)(nBase + n) * ldb + kBase;
  FragU f;
  f.q[0] = *reinterpret_cast<const float4*>(p + hi * 8);
  f.q[1] = *reinterpret_cast<const float4*>(p + 16 + hi * 8);
  return f.v;
}

// A fragment built from an f32 row-major matrix (attention probs), converted
// to f16 in-register.
__device__ __forceinline__ v16h load_frag_a_f32(const float* __restrict__ base,
                                                int lda, int mBase, int kBase) {
  const int lane = threadIdx.x & 31;
  const int m = lane & 15, hi = lane >> 4;
  const float* p = base + (size_t)(mBase + m) * lda + kBase + hi * 8;
  float4 x0 = *reinterpret_cast<const float4*>(p);
  float4 x1 = *reinterpret_cast<const float4*>(p + 4);
  float4 x2 = *reinterpret_cast<const float4*>(p + 16);
  float4 x3 = *reinterpret_cast<const float4*>(p + 20);
  v16h r;
  r[0] = (_Float16)x0.x; r[1] = (_Float16)x0.y; r[2] = (_Float16)x0.z; r[3] = (_Float16)x0.w;
  r[4] = (_Float16)x1.x; r[5] = (_Float16)x1.y; r[6] = (_Float16)x1.z; r[7] = (_Float16)x1.w;
  r[8] = (_Float16)x2.x; r[9] = (_Float16)x2.y; r[10] = (_Float16)x2.z; r[11] = (_Float16)x2.w;
  r[12] = (_Float16)x3.x; r[13] = (_Float16)x3.y; r[14] = (_Float16)x3.z; r[15] = (_Float16)x3.w;
  return r;
}

// ---------------------------------------------------------------------------
// Kernel 0: f32 -> f16 conversion
// ---------------------------------------------------------------------------
__global__ void cvt_f32_to_f16_kernel(const float* __restrict__ src,
                                      _Float16* __restrict__ dst, int n) {
  int i = blockIdx.x * blockDim.x + threadIdx.x;
  if (i < n) dst[i] = (_Float16)src[i];
}

// ---------------------------------------------------------------------------
// Kernel 1: projection GEMM  out[r,o] = sum_i X[r,i] * W[o,i]
// One wave computes a 64x32 tile (4 m-frags x 2 n-frags) => 8 v8f
// accumulators (64 VGPRs), no spills. MODE is a compile-time constant so the
// store epilogue is branch-free.
// MODE 0: write Q as [B,H,S,DK] f16, scaled by 1/sqrt(64)
// MODE 1: write K as [B,H,S,DK] f16
// MODE 2: write V transposed as [B,H,DK,S] f16
// ---------------------------------------------------------------------------
template <int MODE>
__global__ void __launch_bounds__(256, 1)
proj_gemm_kernel(const _Float16* __restrict__ X,
                 const _Float16* __restrict__ W,
                 _Float16* __restrict__ Out) {
  const int wid  = (blockIdx.x * blockDim.x + threadIdx.x) >> 5; // 0..2047
  const int lane = threadIdx.x & 31;
  const int mTile = wid >> 4;       // 0..127  (8192/64)
  const int nTile = wid & 15;       // 0..15   (512/32)
  const int mBase = mTile * 64, nBase = nTile * 32;

  v8f acc[4][2];
#pragma unroll
  for (int i = 0; i < 4; ++i)
#pragma unroll
    for (int j = 0; j < 2; ++j) acc[i][j] = zero_v8f();

  for (int k = 0; k < DM; k += 32) {
    v16h a[4];
#pragma unroll
    for (int i = 0; i < 4; ++i) a[i] = load_frag_a(X, DM, mBase + 16 * i, k);
#pragma unroll
    for (int j = 0; j < 2; ++j) {
      v16h b = load_frag_bT(W, DM, nBase + 16 * j, k);
#pragma unroll
      for (int i = 0; i < 4; ++i) acc[i][j] = wmma16x16x32(a[i], b, acc[i][j]);
    }
  }

  const int hi = lane >> 4, nn = lane & 15;
  constexpr float scale = (MODE == 0) ? 0.125f : 1.0f;  // fold 1/sqrt(d_k) in Q
#pragma unroll
  for (int i = 0; i < 4; ++i)
#pragma unroll
    for (int j = 0; j < 2; ++j)
#pragma unroll
      for (int v = 0; v < 8; ++v) {
        int m = mBase + 16 * i + 8 * hi + v;   // token row
        int n = nBase + 16 * j + nn;           // output feature
        int b = m >> 11, s = m & (SS - 1);
        int h = n >> 6,  d = n & (DK - 1);
        int bh = b * NH + h;
        _Float16 val = (_Float16)(acc[i][j][v] * scale);
        if constexpr (MODE == 2)
          Out[((size_t)bh * DK + d) * SS + s] = val;        // V^T [B,H,DK,S]
        else
          Out[((size_t)bh * SS + s) * DK + d] = val;        // Q/K [B,H,S,DK]
      }
}

// ---------------------------------------------------------------------------
// Kernel 2: fused scores + mask + softmax.
// Block = 256 threads (8 waves) computes a 16(q) x 2048(k) stripe of
// S = Q*K^T for one (b,h); scores staged in 128KB of dynamic LDS (MI455X has
// 320KB/WGP), then masked softmax, single write of attn to global.
// ---------------------------------------------------------------------------
__global__ void scores_softmax_kernel(const _Float16* __restrict__ Qh,
                                      const _Float16* __restrict__ Kh,
                                      const unsigned char* __restrict__ mask,
                                      float* __restrict__ attn_out) {
  extern __shared__ float sm[];            // [16*2048] scores + [256] scratch
  float* red = sm + 16 * SS;

  const int bh = blockIdx.y;               // 0..31
  const int b  = bh >> 3;
  const int qBase = blockIdx.x * 16;
  const _Float16* Q = Qh + (size_t)bh * SS * DK;
  const _Float16* K = Kh + (size_t)bh * SS * DK;
  const int wave = threadIdx.x >> 5;       // 0..7, each covers 256 k-columns
  const int lane = threadIdx.x & 31;
  const int hi = lane >> 4, nn = lane & 15;

  // A fragments shared across all 16 column tiles of this wave (K-dim = 64)
  v16h a0 = load_frag_a(Q, DK, qBase, 0);
  v16h a1 = load_frag_a(Q, DK, qBase, 32);

#pragma unroll 4
  for (int t = 0; t < 16; ++t) {
    int nB = wave * 256 + t * 16;
    v8f c = zero_v8f();
    c = wmma16x16x32(a0, load_frag_bT(K, DK, nB, 0), c);
    c = wmma16x16x32(a1, load_frag_bT(K, DK, nB, 32), c);
    int n = nB + nn;
#pragma unroll
    for (int v = 0; v < 8; ++v) sm[(8 * hi + v) * SS + n] = c[v];
  }
  __syncthreads();

  // Softmax: 16 threads per q-row, 128 columns each.
  const int row  = threadIdx.x >> 4;                 // 0..15
  const int col0 = (threadIdx.x & 15) * 128;
  float* r = sm + row * SS;
  const unsigned char* mrow =
      mask + ((size_t)b * SS + (qBase + row)) * SS;  // bool: 1 byte/elem

  float mx = -3.0e38f;
  for (int c = col0; c < col0 + 128; ++c) {
    float s = r[c];
    if (mrow[c]) s = -1.0e9f;
    r[c] = s;
    mx = fmaxf(mx, s);
  }
  red[threadIdx.x] = mx;
  __syncthreads();
  float rowmax = -3.0e38f;
#pragma unroll
  for (int t = 0; t < 16; ++t) rowmax = fmaxf(rowmax, red[row * 16 + t]);
  __syncthreads();

  float sum = 0.0f;
  for (int c = col0; c < col0 + 128; ++c) {
    float e = __expf(r[c] - rowmax);
    r[c] = e;
    sum += e;
  }
  red[threadIdx.x] = sum;
  __syncthreads();
  float tot = 0.0f;
#pragma unroll
  for (int t = 0; t < 16; ++t) tot += red[row * 16 + t];
  float inv = 1.0f / tot;

  float* orow = attn_out + ((size_t)bh * SS + (qBase + row)) * SS;
  for (int c = col0; c < col0 + 128; ++c) orow[c] = r[c] * inv;
}

// ---------------------------------------------------------------------------
// Kernel 3: prob[q,d] = sum_k attn[q,k] * V[k,d] per (b,h).
// attn read as f32, converted to f16 fragments in-register; V read from the
// transposed [B,H,DK,S] layout (contiguous B-frag loads).
// Wave tile 64(q) x 32(d): 8 accumulators, no spills.
// ---------------------------------------------------------------------------
__global__ void __launch_bounds__(256, 1)
av_gemm_kernel(const float* __restrict__ attn,
               const _Float16* __restrict__ VhT,
               _Float16* __restrict__ Ph16) {
  const int wid  = (blockIdx.x * blockDim.x + threadIdx.x) >> 5; // 0..2047
  const int lane = threadIdx.x & 31;
  const int bh    = wid >> 6;        // 0..31
  const int rem   = wid & 63;
  const int mTile = rem >> 1;        // 0..31  (2048/64)
  const int nTile = rem & 1;         // 0..1   (64/32)
  const int b = bh >> 3, h = bh & 7;
  const int mBase = mTile * 64, dBase = nTile * 32;

  const float*    A = attn + (size_t)bh * SS * SS;
  const _Float16* V = VhT  + (size_t)bh * DK * SS;

  v8f acc[4][2];
#pragma unroll
  for (int i = 0; i < 4; ++i)
#pragma unroll
    for (int j = 0; j < 2; ++j) acc[i][j] = zero_v8f();

  for (int k = 0; k < SS; k += 32) {
    // Latency-hiding prefetch of the next k-step of the streamed attn rows
    // (emits global_prefetch_b8; no counter side effects).
    if (k + 32 < SS) {
      const int lane2 = threadIdx.x & 31;
      const float* pf = A + (size_t)(mBase + (lane2 & 15)) * SS + (k + 32);
      __builtin_prefetch(pf, 0, 1);
    }
    v16h a[4];
#pragma unroll
    for (int i = 0; i < 4; ++i) a[i] = load_frag_a_f32(A, SS, mBase + 16 * i, k);
#pragma unroll
    for (int j = 0; j < 2; ++j) {
      v16h bf = load_frag_bT(V, SS, dBase + 16 * j, k);
#pragma unroll
      for (int i = 0; i < 4; ++i) acc[i][j] = wmma16x16x32(a[i], bf, acc[i][j]);
    }
  }

  const int hi = lane >> 4, nn = lane & 15;
#pragma unroll
  for (int i = 0; i < 4; ++i)
#pragma unroll
    for (int j = 0; j < 2; ++j)
#pragma unroll
      for (int v = 0; v < 8; ++v) {
        int q = mBase + 16 * i + 8 * hi + v;
        int d = dBase + 16 * j + nn;
        Ph16[((size_t)b * SS + q) * DM + h * DK + d] = (_Float16)acc[i][j][v];
      }
}

// ---------------------------------------------------------------------------
// Kernel 4: output projection  O[r,o] = sum_i P[r,i] * W_O[o,i]  (f32 out)
// Wave tile 64x32.
// ---------------------------------------------------------------------------
__global__ void __launch_bounds__(256, 1)
o_gemm_kernel(const _Float16* __restrict__ P,
              const _Float16* __restrict__ WO,
              float* __restrict__ O) {
  const int wid  = (blockIdx.x * blockDim.x + threadIdx.x) >> 5; // 0..2047
  const int lane = threadIdx.x & 31;
  const int mTile = wid >> 4;
  const int nTile = wid & 15;
  const int mBase = mTile * 64, nBase = nTile * 32;

  v8f acc[4][2];
#pragma unroll
  for (int i = 0; i < 4; ++i)
#pragma unroll
    for (int j = 0; j < 2; ++j) acc[i][j] = zero_v8f();

  for (int k = 0; k < DM; k += 32) {
    v16h a[4];
#pragma unroll
    for (int i = 0; i < 4; ++i) a[i] = load_frag_a(P, DM, mBase + 16 * i, k);
#pragma unroll
    for (int j = 0; j < 2; ++j) {
      v16h b = load_frag_bT(WO, DM, nBase + 16 * j, k);
#pragma unroll
      for (int i = 0; i < 4; ++i) acc[i][j] = wmma16x16x32(a[i], b, acc[i][j]);
    }
  }

  const int hi = lane >> 4, nn = lane & 15;
#pragma unroll
  for (int i = 0; i < 4; ++i)
#pragma unroll
    for (int j = 0; j < 2; ++j)
#pragma unroll
      for (int v = 0; v < 8; ++v) {
        int m = mBase + 16 * i + 8 * hi + v;
        int n = nBase + 16 * j + nn;
        O[(size_t)m * DM + n] = acc[i][j][v];
      }
}

// ---------------------------------------------------------------------------
// Kernel 5: residual add + LayerNorm over the 512-wide feature dim.
// One 256-thread block per token row; 2 elements per thread.
// ---------------------------------------------------------------------------
__global__ void add_ln_kernel(const float* __restrict__ X,   // input_Q
                              const float* __restrict__ O,   // projection out
                              const float* __restrict__ gamma,
                              const float* __restrict__ beta,
                              float* __restrict__ out) {
  __shared__ float red[256];
  const int row = blockIdx.x;
  const int t = threadIdx.x;
  const float* x = X + (size_t)row * DM;
  const float* o = O + (size_t)row * DM;

  float v0 = x[t] + o[t];
  float v1 = x[t + 256] + o[t + 256];

  red[t] = v0 + v1;
  __syncthreads();
  for (int s = 128; s > 0; s >>= 1) {
    if (t < s) red[t] += red[t + s];
    __syncthreads();
  }
  float mu = red[0] * (1.0f / DM);
  __syncthreads();

  float d0 = v0 - mu, d1 = v1 - mu;
  red[t] = d0 * d0 + d1 * d1;
  __syncthreads();
  for (int s = 128; s > 0; s >>= 1) {
    if (t < s) red[t] += red[t + s];
    __syncthreads();
  }
  float rstd = rsqrtf(red[0] * (1.0f / DM) + 1e-5f);

  out[(size_t)row * DM + t]       = d0 * rstd * gamma[t] + beta[t];
  out[(size_t)row * DM + t + 256] = d1 * rstd * gamma[t + 256] + beta[t + 256];
}

// ---------------------------------------------------------------------------
extern "C" void kernel_launch(void* const* d_in, const int* in_sizes, int n_in,
                              void* d_out, int out_size, void* d_ws, size_t ws_size,
                              hipStream_t stream) {
  const float* inQ   = (const float*)d_in[0];
  const float* inK   = (const float*)d_in[1];
  const float* inV   = (const float*)d_in[2];
  const unsigned char* mask = (const unsigned char*)d_in[3];  // jnp bool_: 1B
  const float* W_Q   = (const float*)d_in[4];
  const float* W_K   = (const float*)d_in[5];
  const float* W_V   = (const float*)d_in[6];
  const float* W_O   = (const float*)d_in[7];
  const float* gamma = (const float*)d_in[8];
  const float* beta  = (const float*)d_in[9];

  float* out_ln   = (float*)d_out;                       // [4,2048,512]
  float* out_attn = out_ln + (size_t)ROWS * DM;          // [4,8,2048,2048]

  // Workspace carve-up (~78 MB total)
  const size_t NX = (size_t)ROWS * DM;   // 4,194,304
  const size_t NW = (size_t)DM * DM;     // 262,144
  uint8_t* w = (uint8_t*)d_ws;
  _Float16* XQh = (_Float16*)w;                 w += NX * 2;
  _Float16* XKh = (_Float16*)w;                 w += NX * 2;
  _Float16* XVh = (_Float16*)w;                 w += NX * 2;
  _Float16* WQh = (_Float16*)w;                 w += NW * 2;
  _Float16* WKh = (_Float16*)w;                 w += NW * 2;
  _Float16* WVh = (_Float16*)w;                 w += NW * 2;
  _Float16* WOh = (_Float16*)w;                 w += NW * 2;
  _Float16* Qh  = (_Float16*)w;                 w += NX * 2;   // [B,H,S,DK]
  _Float16* Kh  = (_Float16*)w;                 w += NX * 2;   // [B,H,S,DK]
  _Float16* VhT = (_Float16*)w;                 w += NX * 2;   // [B,H,DK,S]
  _Float16* Ph16 = (_Float16*)w;                w += NX * 2;   // [B,S,512]
  float*    O32  = (float*)w;                   w += NX * 4;   // [8192,512]

  // 0) f32 -> f16 conversions
  {
    int thr = 256;
    cvt_f32_to_f16_kernel<<<(int)((NX + thr - 1) / thr), thr, 0, stream>>>(inQ, XQh, (int)NX);
    cvt_f32_to_f16_kernel<<<(int)((NX + thr - 1) / thr), thr, 0, stream>>>(inK, XKh, (int)NX);
    cvt_f32_to_f16_kernel<<<(int)((NX + thr - 1) / thr), thr, 0, stream>>>(inV, XVh, (int)NX);
    cvt_f32_to_f16_kernel<<<(int)((NW + thr - 1) / thr), thr, 0, stream>>>(W_Q, WQh, (int)NW);
    cvt_f32_to_f16_kernel<<<(int)((NW + thr - 1) / thr), thr, 0, stream>>>(W_K, WKh, (int)NW);
    cvt_f32_to_f16_kernel<<<(int)((NW + thr - 1) / thr), thr, 0, stream>>>(W_V, WVh, (int)NW);
    cvt_f32_to_f16_kernel<<<(int)((NW + thr - 1) / thr), thr, 0, stream>>>(W_O, WOh, (int)NW);
  }

  // 1) Q/K/V projections: 2048 wave-tiles -> 256 blocks x 8 waves
  proj_gemm_kernel<0><<<256, 256, 0, stream>>>(XQh, WQh, Qh);
  proj_gemm_kernel<1><<<256, 256, 0, stream>>>(XKh, WKh, Kh);
  proj_gemm_kernel<2><<<256, 256, 0, stream>>>(XVh, WVh, VhT);

  // 2) fused scores + mask + softmax (128KB dynamic LDS per workgroup)
  {
    size_t smem = (size_t)(16 * SS + 256) * sizeof(float);   // 132,096 B
    (void)hipFuncSetAttribute(
        reinterpret_cast<const void*>(scores_softmax_kernel),
        hipFuncAttributeMaxDynamicSharedMemorySize, (int)smem);
    scores_softmax_kernel<<<dim3(SS / 16, BB * NH), 256, smem, stream>>>(
        Qh, Kh, mask, out_attn);
  }

  // 3) attn @ V  -> Ph16 [B,S,512] f16
  av_gemm_kernel<<<256, 256, 0, stream>>>(out_attn, VhT, Ph16);

  // 4) output projection -> O32 f32
  o_gemm_kernel<<<256, 256, 0, stream>>>(Ph16, WOh, O32);

  // 5) residual + LayerNorm -> d_out part 1
  add_ln_kernel<<<ROWS, 256, 0, stream>>>(inQ, O32, gamma, beta, out_ln);
}
